// DCN_48996986913013
// MI455X (gfx1250) — compile-verified, matrix operand order checked by reading, not run on hardware
//
#include <hip/hip_runtime.h>
#include <hip/hip_bf16.h>
#include <math.h>

typedef __attribute__((ext_vector_type(16))) _Float16 v16h;
typedef __attribute__((ext_vector_type(8)))  float    v8f;

// ---------------------------------------------------------------------------
// 16-bit A/B WMMA fragment element -> K mapping (ISA 7.12.2):
// lanes 0-15: K base 0, lanes 16-31: K base 8.
// elems 0..7 -> K base+0..7 ; elems 8..15 -> K base+16..23.
// ---------------------------------------------------------------------------
__device__ __forceinline__ int kOfElem(int lane, int j) {
    int base = (lane < 16) ? 0 : 8;
    int v = j >> 1, p = j & 1;
    return base + ((v < 4) ? (2 * v + p) : (16 + 2 * (v - 4) + p));
}

// Inverse: (pixel p, reduction index k) -> flat index in fragment-ordered
// storage [tile][kt][lane][16].  K block 0:[0-7]->half0, 1:[8-15]->half1,
// 2:[16-23]->half0/j+8, 3:[24-31]->half1/j+8.
template<int KT>
__device__ __forceinline__ int fragIdx(int p, int k) {
    const int tile = p >> 4, m = p & 15;
    const int kt = k >> 5, ko = k & 31;
    const int block = ko >> 3, within = ko & 7;
    const int lane = ((block & 1) << 4) + m;
    const int j = ((block >> 1) << 3) + within;
    return ((tile * KT + kt) * 32 + lane) * 16 + j;
}

// ---------------------------------------------------------------------------
// Pack f32 weights (OC, IC, 3, 3) into f16 B-fragments:
// wp[((nt*KT + kt)*32 + lane)*16 + j] = w[n=nt*16+lane%16][k=kt*32+kOfElem]
// with zero padding for n >= OC or k >= IC*9.
// ---------------------------------------------------------------------------
template<int IC, int OC, int KPAD, int NT>
__global__ __launch_bounds__(256)
void pack_w(const float* __restrict__ w, _Float16* __restrict__ wp)
{
    constexpr int KT = KPAD / 32;
    const int total = NT * KT * 32 * 16;
    for (int idx = blockIdx.x * 256 + threadIdx.x; idx < total; idx += gridDim.x * 256) {
        const int j    = idx & 15;
        const int lane = (idx >> 4) & 31;
        const int kt   = (idx >> 9) % KT;
        const int nt   = idx / (KT * 512);
        const int n    = nt * 16 + (lane & 15);
        const int kidx = kt * 32 + kOfElem(lane, j);
        const float wv = (n < OC && kidx < IC * 9) ? w[n * IC * 9 + kidx] : 0.f;
        wp[idx] = (_Float16)wv;
    }
}

// ---------------------------------------------------------------------------
// Core GEMM + epilogue.  AsF holds the im2col tile in A-fragment order;
// wp holds packed weights.  Per k-step: 32B ds load (A) + NT x 32B vmem (B)
// + NT x v_wmma.  sBias is a padded NT*16 LDS array (zeros if no bias).
// ---------------------------------------------------------------------------
template<int OC, int KPAD, int NT, bool RELU, bool STATS>
__device__ __forceinline__ void wmma_gemm_store(const _Float16* AsF, const _Float16* __restrict__ wp,
                                                const float* sBias, float* __restrict__ out,
                                                int b, int y, float* sSum, float* sSq)
{
    constexpr int KT = KPAD / 32;
    const int lane  = threadIdx.x & 31;
    const int wave  = threadIdx.x >> 5;
    const int mrow  = lane & 15;
    const int mbase = (lane < 16) ? 0 : 8;

    v8f acc[NT];
#pragma unroll
    for (int nt = 0; nt < NT; ++nt) {
        v8f z = {0.f, 0.f, 0.f, 0.f, 0.f, 0.f, 0.f, 0.f};
        acc[nt] = z;
    }

    const _Float16* aBase = AsF + wave * (KT * 32 * 16);
    for (int kt = 0; kt < KT; ++kt) {
        const v16h afrag = *(const v16h*)(aBase + (kt * 32 + lane) * 16);
#pragma unroll
        for (int nt = 0; nt < NT; ++nt) {
            const v16h bfrag = *(const v16h*)(wp + ((nt * KT + kt) * 32 + lane) * 16);
            acc[nt] = __builtin_amdgcn_wmma_f32_16x16x32_f16(
                false, afrag, false, bfrag, (short)0, acc[nt], false, false);
        }
    }

#pragma unroll
    for (int nt = 0; nt < NT; ++nt) {
        const int n = nt * 16 + mrow;
        const float bv = sBias[nt * 16 + mrow];
        float vals[8];
        float lsum = 0.f, lsq = 0.f;
#pragma unroll
        for (int r = 0; r < 8; ++r) {
            float v = acc[nt][r] + bv;
            if (RELU) v = fmaxf(v, 0.f);
            vals[r] = v;
            lsum += v;
            lsq  += v * v;
        }
        if (n < OC) {
            float* o = out + ((b * OC + n) * 64 + y) * 64 + wave * 16 + mbase;
            float4 v0 = {vals[0], vals[1], vals[2], vals[3]};
            float4 v1 = {vals[4], vals[5], vals[6], vals[7]};
            *(float4*)(o)     = v0;
            *(float4*)(o + 4) = v1;
            if (STATS) {
                atomicAdd(&sSum[n], lsum);
                atomicAdd(&sSq[n],  lsq);
            }
        }
    }
}

// ---------------------------------------------------------------------------
// Regular 3x3 conv (pad 1), folded BN affine on input, optional bias/ReLU/
// BN-stat accumulation.  Grid: B*H blocks of 128 threads (one output row).
// ---------------------------------------------------------------------------
template<int IC, int OC, int KPAD, int NT, bool RELU, bool STATS, bool BIAS>
__global__ __launch_bounds__(128)
void conv3x3_wmma(const float* __restrict__ in, const _Float16* __restrict__ wp,
                  const float* __restrict__ bias,
                  const float* __restrict__ scale, const float* __restrict__ shift,
                  float* __restrict__ out, float* __restrict__ gsum, float* __restrict__ gsq)
{
    constexpr int KT = KPAD / 32;
    __shared__ __align__(32) _Float16 AsF[64 * KPAD];
    __shared__ float sSum[OC];
    __shared__ float sSq[OC];
    __shared__ float sBias[NT * 16];
    __shared__ float lsc[IC];
    __shared__ float lsh[IC];

    const int b = blockIdx.x >> 6;
    const int y = blockIdx.x & 63;
    const int tid = threadIdx.x;

    if (tid < IC) {
        lsc[tid] = scale ? scale[tid] : 1.f;
        lsh[tid] = shift ? shift[tid] : 0.f;
    }
    if (tid < NT * 16) sBias[tid] = (BIAS && tid < OC) ? bias[tid] : 0.f;
    if (STATS && tid < OC) { sSum[tid] = 0.f; sSq[tid] = 0.f; }
    __syncthreads();

    for (int idx = tid; idx < 64 * KPAD; idx += 128) {
        const int p = idx / KPAD;
        const int k = idx - p * KPAD;
        float v = 0.f;
        if (k < IC * 9) {
            const int ic = k / 9;
            const int kk = k - ic * 9;
            const int yy = y + kk / 3 - 1;
            const int xx = p + kk % 3 - 1;
            if (yy >= 0 && yy < 64 && xx >= 0 && xx < 64)
                v = fmaf(in[((b * IC + ic) * 64 + yy) * 64 + xx], lsc[ic], lsh[ic]);
        }
        AsF[fragIdx<KT>(p, k)] = (_Float16)v;
    }
    __syncthreads();

    wmma_gemm_store<OC, KPAD, NT, RELU, STATS>(AsF, wp, sBias, out, b, y, sSum, sSq);

    if (STATS) {
        __syncthreads();
        if (tid < OC) {
            atomicAdd(&gsum[tid], sSum[tid]);
            atomicAdd(&gsq[tid],  sSq[tid]);
        }
    }
}

// ---------------------------------------------------------------------------
// Deformable 3x3 conv: bilinear-sampled im2col tile (fragment order), then
// the same WMMA GEMM.  Offsets (B, 9, 2, H, W); zero padding outside image.
// ---------------------------------------------------------------------------
template<int IC, int OC, int KPAD, int NT, bool RELU, bool STATS>
__global__ __launch_bounds__(128)
void dconv3x3_wmma(const float* __restrict__ in, const float* __restrict__ off,
                   const _Float16* __restrict__ wp,
                   const float* __restrict__ scale, const float* __restrict__ shift,
                   float* __restrict__ out, float* __restrict__ gsum, float* __restrict__ gsq)
{
    constexpr int KT = KPAD / 32;
    __shared__ __align__(32) _Float16 AsF[64 * KPAD];
    __shared__ float sSum[OC];
    __shared__ float sSq[OC];
    __shared__ float sBias[NT * 16];
    __shared__ float lsc[IC];
    __shared__ float lsh[IC];

    const int b = blockIdx.x >> 6;
    const int y = blockIdx.x & 63;
    const int tid = threadIdx.x;

    if (tid < IC) { lsc[tid] = scale[tid]; lsh[tid] = shift[tid]; }
    if (tid < NT * 16) sBias[tid] = 0.f;
    if (STATS && tid < OC) { sSum[tid] = 0.f; sSq[tid] = 0.f; }

    for (int idx = tid; idx < 64 * KPAD; idx += 128)
        AsF[idx] = (_Float16)0.f;
    __syncthreads();

    for (int t = tid; t < 64 * 9; t += 128) {
        const int p  = t / 9;
        const int kk = t - p * 9;
        const int ky = kk / 3, kx = kk % 3;
        const float offy = off[((b * 18 + kk * 2 + 0) * 64 + y) * 64 + p];
        const float offx = off[((b * 18 + kk * 2 + 1) * 64 + y) * 64 + p];
        const float py = (float)y + (float)(ky - 1) + offy;
        const float px = (float)p + (float)(kx - 1) + offx;
        const float y0f = floorf(py), x0f = floorf(px);
        const float wy = py - y0f, wx = px - x0f;
        const int y0 = (int)y0f, x0 = (int)x0f;
        const int y1 = y0 + 1,   x1 = x0 + 1;
        const bool vy0 = (y0 >= 0 && y0 < 64), vy1 = (y1 >= 0 && y1 < 64);
        const bool vx0 = (x0 >= 0 && x0 < 64), vx1 = (x1 >= 0 && x1 < 64);
        const int cy0 = min(max(y0, 0), 63), cy1 = min(max(y1, 0), 63);
        const int cx0 = min(max(x0, 0), 63), cx1 = min(max(x1, 0), 63);
        const float w00 = (1.f - wy) * (1.f - wx), w01 = (1.f - wy) * wx;
        const float w10 = wy * (1.f - wx),         w11 = wy * wx;
        for (int c = 0; c < IC; ++c) {
            const float* img = in + ((size_t)(b * IC + c)) * 4096;
            const float s = lsc[c], sh = lsh[c];
            const float v00 = (vy0 && vx0) ? fmaf(img[cy0 * 64 + cx0], s, sh) : 0.f;
            const float v01 = (vy0 && vx1) ? fmaf(img[cy0 * 64 + cx1], s, sh) : 0.f;
            const float v10 = (vy1 && vx0) ? fmaf(img[cy1 * 64 + cx0], s, sh) : 0.f;
            const float v11 = (vy1 && vx1) ? fmaf(img[cy1 * 64 + cx1], s, sh) : 0.f;
            AsF[fragIdx<KT>(p, c * 9 + kk)] =
                (_Float16)(v00 * w00 + v01 * w01 + v10 * w10 + v11 * w11);
        }
    }
    __syncthreads();

    wmma_gemm_store<OC, KPAD, NT, RELU, STATS>(AsF, wp, sBias, out, b, y, sSum, sSq);

    if (STATS) {
        __syncthreads();
        if (tid < OC) {
            atomicAdd(&gsum[tid], sSum[tid]);
            atomicAdd(&gsq[tid],  sSq[tid]);
        }
    }
}

// ---------------------------------------------------------------------------
__global__ __launch_bounds__(64)
void bn_stats(const float* __restrict__ sum, const float* __restrict__ sq,
              const float* __restrict__ g, const float* __restrict__ bb,
              float* __restrict__ scale, float* __restrict__ shift, int C, float invN)
{
    const int c = threadIdx.x;
    if (c >= C) return;
    const float m  = sum[c] * invN;
    const float v  = sq[c] * invN - m * m;
    const float sc = g[c] * rsqrtf(v + 1e-5f);
    scale[c] = sc;
    shift[c] = bb[c] - m * sc;
}

// ---------------------------------------------------------------------------
__global__ __launch_bounds__(256)
void norm_pool(float* __restrict__ em, const float* __restrict__ scale,
               const float* __restrict__ shift, float* __restrict__ pool)
{
    const int b = blockIdx.x >> 6;
    const int c = blockIdx.x & 63;
    __shared__ float q[4];
    if (threadIdx.x < 4) q[threadIdx.x] = 0.f;
    __syncthreads();

    const float s = scale[c], sh = shift[c];
    float* base = em + ((size_t)(b * 64 + c)) * 4096;
    float local[4] = {0.f, 0.f, 0.f, 0.f};
    for (int i = threadIdx.x; i < 4096; i += 256) {
        const int yy = i >> 6, xx = i & 63;
        const float v = fmaf(base[i], s, sh);
        base[i] = v;
        local[((yy >= 32) ? 2 : 0) + ((xx >= 32) ? 1 : 0)] += v;
    }
#pragma unroll
    for (int k = 0; k < 4; ++k) atomicAdd(&q[k], local[k]);
    __syncthreads();
    if (threadIdx.x < 4)
        pool[(b * 64 + c) * 4 + threadIdx.x] = q[threadIdx.x] * (1.0f / 1024.0f);
}

// ---------------------------------------------------------------------------
__global__ __launch_bounds__(64)
void fc_softmax(const float* __restrict__ pool, const float* __restrict__ fw,
                const float* __restrict__ fb, float* __restrict__ prob)
{
    const int b = threadIdx.x;
    const float* p = pool + b * 256;
    float logits[10];
#pragma unroll
    for (int o = 0; o < 10; ++o) {
        float acc = fb[o];
        for (int j = 0; j < 256; ++j) acc += p[j] * fw[o * 256 + j];
        logits[o] = acc;
    }
    float mx = logits[0];
#pragma unroll
    for (int o = 1; o < 10; ++o) mx = fmaxf(mx, logits[o]);
    float denom = 0.f;
#pragma unroll
    for (int o = 0; o < 10; ++o) { logits[o] = expf(logits[o] - mx); denom += logits[o]; }
    const float inv = 1.0f / denom;
#pragma unroll
    for (int o = 0; o < 10; ++o) prob[b * 10 + o] = logits[o] * inv;
}

// ---------------------------------------------------------------------------
extern "C" void kernel_launch(void* const* d_in, const int* in_sizes, int n_in,
                              void* d_out, int out_size, void* d_ws, size_t ws_size,
                              hipStream_t stream)
{
    const float* x   = (const float*)d_in[0];
    const float* w1  = (const float*)d_in[1];
    const float* g1  = (const float*)d_in[2];
    const float* b1  = (const float*)d_in[3];
    const float* w2  = (const float*)d_in[4];
    const float* bc2 = (const float*)d_in[5];
    const float* g2  = (const float*)d_in[6];
    const float* b2  = (const float*)d_in[7];
    const float* w3  = (const float*)d_in[8];
    const float* wd1 = (const float*)d_in[9];
    const float* g3  = (const float*)d_in[10];
    const float* b3  = (const float*)d_in[11];
    const float* w4  = (const float*)d_in[12];
    const float* bc4 = (const float*)d_in[13];
    const float* wd2 = (const float*)d_in[14];
    const float* g4  = (const float*)d_in[15];
    const float* b4  = (const float*)d_in[16];
    const float* fw  = (const float*)d_in[17];
    const float* fb  = (const float*)d_in[18];

    float* ws    = (float*)d_ws;
    float* a1    = ws;                    // 64*8*64*64   = 2,097,152
    float* a2    = a1   + 2097152;        // 64*20*64*64  = 5,242,880
    float* offb  = a2   + 5242880;        // 64*18*64*64  = 4,718,592 (off1 then off2)
    float* a3    = offb + 4718592;        // 64*40*64*64  = 10,485,760
    float* stats = a3   + 10485760;       // 1024 floats
    float* pool  = stats + 1024;          // 16,384 floats

    float* sum0 = stats +   0, *sum1 = stats +  64, *sum2 = stats + 128, *sum3 = stats + 192;
    float* sq0  = stats + 256, *sq1  = stats + 320, *sq2  = stats + 384, *sq3  = stats + 448;
    float* sc0  = stats + 512, *sc1  = stats + 576, *sc2  = stats + 640, *sc3  = stats + 704;
    float* sh0  = stats + 768, *sh1  = stats + 832, *sh2  = stats + 896, *sh3  = stats + 960;

    // packed f16 weights (fragment order), 32B-aligned region after pool
    _Float16* wp1  = (_Float16*)(pool + 16384);  // 1*1*512   = 512
    _Float16* wp2  = wp1  + 512;                 // 2*3*512   = 3072
    _Float16* wpo1 = wp2  + 3072;                // 2*6*512   = 6144
    _Float16* wpd1 = wpo1 + 6144;                // 3*6*512   = 9216
    _Float16* wpo2 = wpd1 + 9216;                // 2*12*512  = 12288
    _Float16* wpd2 = wpo2 + 12288;               // 4*12*512  = 24576

    float* prob = (float*)d_out;
    float* em   = (float*)d_out + 640;

    const float invN = 1.0f / (64.0f * 64.0f * 64.0f);
    const dim3 grid(64 * 64), blk(128);

    hipMemsetAsync(stats, 0, 512 * sizeof(float), stream);

    // pack all weights into B-fragment layout (tiny)
    pack_w<3,  8,  32,  1><<<2,  256, 0, stream>>>(w1,  wp1);
    pack_w<8,  20, 96,  2><<<12, 256, 0, stream>>>(w2,  wp2);
    pack_w<20, 18, 192, 2><<<24, 256, 0, stream>>>(w3,  wpo1);
    pack_w<20, 40, 192, 3><<<36, 256, 0, stream>>>(wd1, wpd1);
    pack_w<40, 18, 384, 2><<<48, 256, 0, stream>>>(w4,  wpo2);
    pack_w<40, 64, 384, 4><<<96, 256, 0, stream>>>(wd2, wpd2);

    // layer 1: conv 3->8, relu, bn-stats
    conv3x3_wmma<3, 8, 32, 1, true, true, false><<<grid, blk, 0, stream>>>(
        x, wp1, nullptr, nullptr, nullptr, a1, sum0, sq0);
    bn_stats<<<1, 64, 0, stream>>>(sum0, sq0, g1, b1, sc0, sh0, 8, invN);

    // layer 2: conv 8->20 (+bias), relu, bn-stats (bn1 folded into load)
    conv3x3_wmma<8, 20, 96, 2, true, true, true><<<grid, blk, 0, stream>>>(
        a1, wp2, bc2, sc0, sh0, a2, sum1, sq1);
    bn_stats<<<1, 64, 0, stream>>>(sum1, sq1, g2, b2, sc1, sh1, 20, invN);

    // offsets 1: conv 20->18 (bn2 folded)
    conv3x3_wmma<20, 18, 192, 2, false, false, false><<<grid, blk, 0, stream>>>(
        a2, wpo1, nullptr, sc1, sh1, offb, nullptr, nullptr);

    // deform conv 1: 20->40, relu, bn-stats
    dconv3x3_wmma<20, 40, 192, 3, true, true><<<grid, blk, 0, stream>>>(
        a2, offb, wpd1, sc1, sh1, a3, sum2, sq2);
    bn_stats<<<1, 64, 0, stream>>>(sum2, sq2, g3, b3, sc2, sh2, 40, invN);

    // offsets 2: conv 40->18 (+bias, bn3 folded)
    conv3x3_wmma<40, 18, 384, 2, false, false, true><<<grid, blk, 0, stream>>>(
        a3, wpo2, bc4, sc2, sh2, offb, nullptr, nullptr);

    // deform conv 2: 40->64, relu, bn-stats; writes raw em directly to d_out
    dconv3x3_wmma<40, 64, 384, 4, true, true><<<grid, blk, 0, stream>>>(
        a3, offb, wpd2, sc2, sh2, em, sum3, sq3);
    bn_stats<<<1, 64, 0, stream>>>(sum3, sq3, g4, b4, sc3, sh3, 64, invN);

    // final bn applied in place + quadrant pooling
    norm_pool<<<4096, 256, 0, stream>>>(em, sc3, sh3, pool);

    // classifier head
    fc_softmax<<<1, 64, 0, stream>>>(pool, fw, fb, prob);
}
